// ChemicalLLM_9921374454210
// MI455X (gfx1250) — matrix operand order repaired and checked by reading, not compiled, not run on hardware
//
#include <hip/hip_runtime.h>

typedef _Float16 hvec8  __attribute__((ext_vector_type(8)));
typedef _Float16 hvec16 __attribute__((ext_vector_type(16)));
typedef float    fvec8  __attribute__((ext_vector_type(8)));

constexpr int Bn = 8, Sn = 1024, Nn = 64, Vn = 32000;

// Dynamic-LDS layout for the recurrence kernel (bytes):
//   xh    f16[16*64]   @ 0      (2048)
//   xs    f32[16*64]   @ 2048   (4096)
//   inter f32[16*64]   @ 6144   (4096)
//   hcur  f32[8*64]    @ 10240  (2048)
//   ssum  f32[8]       @ 12288  (32)
//   wbuf  f32[2][512]  @ 12320  (4096)   TDM double buffer for word slabs
//   Wl    f16 frags    @ 16416  (294912) = 16 waves * 18 frags * 1KB
constexpr int LDS_XH = 0, LDS_XS = 2048, LDS_IN = 6144, LDS_HC = 10240,
              LDS_SS = 12288, LDS_WB = 12320, LDS_WL = 16416;
constexpr int LDS_TOTAL = 16416 + 16 * 18 * 1024;   // 311,328 B (< 320KB/WG on CDNA5)

#if __has_builtin(__builtin_amdgcn_tensor_load_to_lds) && __has_builtin(__builtin_amdgcn_s_wait_tensorcnt)
#define HAVE_TDM 1
typedef unsigned int u32x4 __attribute__((ext_vector_type(4)));
typedef int          i32x4 __attribute__((ext_vector_type(4)));
typedef int          i32x8 __attribute__((ext_vector_type(8)));

// Issue a TDM load of one 2KB word slab (512 f32, 1 row) into LDS at lds_off.
// D# per ISA 8.3/8.4: g0 = {count=1, lds_addr, global_addr, type=2};
// g1 = {data_size=4B, tensor_dim0=512, tensor_dim1=1, tile_dim0=512,
//       tile_dim1=1, tensor_dim0_stride=512}.  Groups 2/3 unused (2D).
__device__ __forceinline__ void tdm_load_word(const float* gsrc, unsigned lds_off) {
  unsigned long long ga = (unsigned long long)(size_t)gsrc;
  u32x4 g0;
  g0[0] = 1u;                               // count=1 valid user descriptor
  g0[1] = lds_off;                          // lds_addr (bytes)
  g0[2] = (unsigned)ga;                     // global_addr[31:0]
  g0[3] = ((unsigned)(ga >> 32) & 0x1FFFFFFu) | 0x80000000u;  // ga[56:32] | type=2
  i32x8 g1;
  g1[0] = 2 << 16;                          // data_size = 2 (4 bytes)
  g1[1] = (512 & 0xFFFF) << 16;             // tensor_dim0 lo16 (bits 63:48)
  g1[2] = 1 << 16;                          // tensor_dim0 hi16=0 | tensor_dim1 lo16=1
  g1[3] = 512 << 16;                        // tensor_dim1 hi16=0 | tile_dim0=512
  g1[4] = 1;                                // tile_dim1=1, tile_dim2=0
  g1[5] = 512;                              // tensor_dim0_stride lo32
  g1[6] = 0;
  g1[7] = 0;
#if __clang_major__ >= 23
  __builtin_amdgcn_tensor_load_to_lds(g0, g1, (i32x4)0, (i32x4)0, (i32x8)0, 0);
#else
  __builtin_amdgcn_tensor_load_to_lds(g0, g1, (i32x4)0, (i32x4)0, 0);
#endif
}
#endif

// ---------------------------------------------------------------------------
// Pack W[64][64][64] f32 into f16 WMMA A-fragments for the stage-A GEMM:
// rows r=(i*64+k) (M=4096), cols j (K=64).  Fragment f = mt*2+kf.
// Lane layout (16-bit A 16x32, contiguous K): lane<16 holds K{0..15},
// lane>=16 holds K{16..31}; stored lane-contiguous (32B per lane).
// ---------------------------------------------------------------------------
__global__ void pack_W(const float* __restrict__ W, _Float16* __restrict__ Wp) {
  int t = blockIdx.x * blockDim.x + threadIdx.x;   // 0..16383  (512 frags * 32 lanes)
  int f  = t >> 5;
  int l  = t & 31;
  int mt = f >> 1, kf = f & 1;
  int r  = mt * 16 + (l & 15);
  int i  = r >> 6, k = r & 63;
  int lo = (l < 16) ? 0 : 16;
  _Float16* dst = Wp + (size_t)t * 16;
#pragma unroll
  for (int h = 0; h < 16; ++h) {
    int jl = lo + h;                               // {0..15} or {16..31}
    int j  = kf * 32 + jl;
    dst[h] = (_Float16)W[((size_t)i * 64 + j) * 64 + k];
  }
}

__global__ void cvt_f16(const float* __restrict__ src, _Float16* __restrict__ dst, int n) {
  int i = blockIdx.x * blockDim.x + threadIdx.x;
  if (i < n) dst[i] = (_Float16)src[i];
}

// word[s][b][n] = relu(emb[ids[b][s]][n])
__global__ void embed_relu(const int* __restrict__ ids, const float* __restrict__ emb,
                           float* __restrict__ word) {
  int idx = blockIdx.x * 256 + threadIdx.x;        // 0..524287 = (s*8+b)*64+n
  int n = idx & 63, b = (idx >> 6) & 7, s = idx >> 9;
  int id = ids[b * Sn + s];
  float v = emb[(size_t)id * 64 + n];
  word[idx] = v > 0.f ? v : 0.f;
}

// ---------------------------------------------------------------------------
// Sequential scan, single workgroup, 512 threads = 16 waves.  Wave w owns
// M-tiles mt in [16w,16w+16): tiles 0..6 register-resident (112 VGPRs of A),
// tiles 7..15 streamed from LDS (288KB staged once).  word slabs arrive via
// TDM (tensor_load_to_lds) double-buffered one step ahead.  Per step:
//   x = 0.9h + word;  Y[4096x8] = Wflat * X^T  (v_wmma_f32_16x16x32_f16);
//   inter[b,k] += x[b,i]*Y  (per-lane scale + LDS atomic reduce);
//   h = relu(x + 0.2*inter)/(sum+eps);  stored f16 for the decode GEMM.
// ---------------------------------------------------------------------------
__global__ void __launch_bounds__(512, 1)
recurrence(const _Float16* __restrict__ Wp, const float* __restrict__ word,
           _Float16* __restrict__ Hh, float* __restrict__ hT) {
  extern __shared__ __align__(32) char smem[];
  _Float16* xh    = (_Float16*)(smem + LDS_XH);
  float*    xs    = (float*)(smem + LDS_XS);
  float*    inter = (float*)(smem + LDS_IN);
  float*    hcur  = (float*)(smem + LDS_HC);
  float*    ssum  = (float*)(smem + LDS_SS);
  float*    wbuf  = (float*)(smem + LDS_WB);
  _Float16* Wl    = (_Float16*)(smem + LDS_WL);

  const int t = threadIdx.x;
  const int lane = t & 31;
  const int w = t >> 5;                            // 0..15
  const int bcol = lane & 15;                      // WMMA N column = batch
  const int hi = (lane & 16) ? 1 : 0;
  const int ldsoff = bcol * 64 + (hi ? 16 : 0);    // B frag: contiguous K run per half-wave

  // Register-resident tiles tt=0..6 of this wave (14 fragments, 112 VGPRs)
  hvec16 afr[14];
#pragma unroll
  for (int tt = 0; tt < 7; ++tt) {
    int mt = w * 16 + tt;
    const _Float16* pa = Wp + ((size_t)(mt * 2) * 32 + lane) * 16;
    afr[2 * tt]     = *(const hvec16*)pa;          // kf = 0
    afr[2 * tt + 1] = *(const hvec16*)(pa + 512);  // kf = 1
  }
  // LDS-staged tiles tt=7..15 -> Wl, 1KB per fragment, lane-contiguous
#pragma unroll
  for (int sidx = 0; sidx < 9; ++sidx) {
    int mt = w * 16 + 7 + sidx;
    const _Float16* pa = Wp + ((size_t)(mt * 2) * 32 + lane) * 16;
    int fl = (w * 9 + sidx) * 2;                   // local fragment index
    *(hvec16*)&Wl[fl * 512 + lane * 16]       = *(const hvec16*)pa;
    *(hvec16*)&Wl[(fl + 1) * 512 + lane * 16] = *(const hvec16*)(pa + 512);
  }

  for (int idx = t; idx < 16 * 64; idx += 512) { xh[idx] = (_Float16)0.f; xs[idx] = 0.f; }
  for (int idx = t; idx < 8 * 64; idx += 512) hcur[idx] = 0.f;

#if HAVE_TDM
  if (w == 0) {                                    // prime the pipeline: slab 0
    tdm_load_word(word, LDS_WB);
    __builtin_amdgcn_s_wait_tensorcnt(0);
  }
#endif
  __syncthreads();

  for (int st = 0; st < Sn; ++st) {
#if HAVE_TDM
    if (w == 0 && st + 1 < Sn)                     // fetch next slab during this step
      tdm_load_word(word + (size_t)(st + 1) * 512, LDS_WB + (((st + 1) & 1) << 11));
#endif
    // phase 1: x = (1-DECAY)h + word_t ; stage f16 copy ; zero inter
    {
      int idx = t;
#if HAVE_TDM
      float wv = wbuf[((st & 1) << 9) + idx];
#else
      float wv = word[(size_t)st * 512 + idx];
      if (st + 1 < Sn) __builtin_prefetch(word + (size_t)(st + 1) * 512 + idx, 0, 1);
#endif
      float x = 0.9f * hcur[idx] + wv;
      xs[idx] = x;
      xh[idx] = (_Float16)x;
      inter[idx] = 0.f;
      inter[idx + 512] = 0.f;
    }
    __syncthreads();

    hvec16 b0 = *(const hvec16*)&xh[ldsoff];        // K = j 0..31
    hvec16 b1 = *(const hvec16*)&xh[ldsoff + 32];   // K = j 32..63

    float acc[4][8];
#pragma unroll
    for (int kg = 0; kg < 4; ++kg)
#pragma unroll
      for (int r = 0; r < 8; ++r) acc[kg][r] = 0.f;

#pragma unroll
    for (int p = 0; p < 7; ++p) {
      // streamed tile (tt = 7+p): issue LDS loads early
      int fl = (w * 9 + p) * 2;
      hvec16 s0 = *(const hvec16*)&Wl[fl * 512 + lane * 16];
      hvec16 s1 = *(const hvec16*)&Wl[(fl + 1) * 512 + lane * 16];

      // register-resident tile (tt = p): no load dependencies
      {
        fvec8 d = {0.f, 0.f, 0.f, 0.f, 0.f, 0.f, 0.f, 0.f};
        d = __builtin_amdgcn_wmma_f32_16x16x32_f16(false, afr[2 * p],     false, b0, (short)0, d, false, false);
        d = __builtin_amdgcn_wmma_f32_16x16x32_f16(false, afr[2 * p + 1], false, b1, (short)0, d, false, false);
        int mt = w * 16 + p;
        int i = mt >> 2, kg = mt & 3;
        float xv = xs[bcol * 64 + i];
#pragma unroll
        for (int r = 0; r < 8; ++r) acc[kg][r] += xv * d[r];
      }
      // streamed tile compute
      {
        fvec8 e = {0.f, 0.f, 0.f, 0.f, 0.f, 0.f, 0.f, 0.f};
        e = __builtin_amdgcn_wmma_f32_16x16x32_f16(false, s0, false, b0, (short)0, e, false, false);
        e = __builtin_amdgcn_wmma_f32_16x16x32_f16(false, s1, false, b1, (short)0, e, false, false);
        int mt = w * 16 + 7 + p;
        int i = mt >> 2, kg = mt & 3;
        float xv = xs[bcol * 64 + i];
#pragma unroll
        for (int r = 0; r < 8; ++r) acc[kg][r] += xv * e[r];
      }
    }
#pragma unroll
    for (int q = 7; q < 9; ++q) {                  // remaining streamed tiles tt=14,15
      int fl = (w * 9 + q) * 2;
      hvec16 s0 = *(const hvec16*)&Wl[fl * 512 + lane * 16];
      hvec16 s1 = *(const hvec16*)&Wl[(fl + 1) * 512 + lane * 16];
      fvec8 e = {0.f, 0.f, 0.f, 0.f, 0.f, 0.f, 0.f, 0.f};
      e = __builtin_amdgcn_wmma_f32_16x16x32_f16(false, s0, false, b0, (short)0, e, false, false);
      e = __builtin_amdgcn_wmma_f32_16x16x32_f16(false, s1, false, b1, (short)0, e, false, false);
      int mt = w * 16 + 7 + q;
      int i = mt >> 2, kg = mt & 3;
      float xv = xs[bcol * 64 + i];
#pragma unroll
      for (int r = 0; r < 8; ++r) acc[kg][r] += xv * e[r];
    }

    if (bcol < 8) {
      int krow = hi ? 8 : 0;
#pragma unroll
      for (int kg = 0; kg < 4; ++kg)
#pragma unroll
        for (int r = 0; r < 8; ++r)
          atomicAdd(&inter[bcol * 64 + kg * 16 + krow + r], acc[kg][r]);
    }
    __syncthreads();

    // phase 3: activation + normalization
    {
      int idx = t;
      float xa = xs[idx] + 0.2f * inter[idx];
      inter[idx] = xa > 0.f ? xa : 0.f;
    }
    __syncthreads();
    if (t < 8) {
      float s0 = 0.f;
#pragma unroll
      for (int n = 0; n < 64; ++n) s0 += inter[t * 64 + n];
      ssum[t] = s0 + 1e-8f;
    }
    __syncthreads();
    {
      int idx = t;
      float hv = inter[idx] / ssum[idx >> 6];
      hcur[idx] = hv;
      int b = idx >> 6, n = idx & 63;
      Hh[((size_t)b * Sn + st) * 64 + n] = (_Float16)hv;   // row r = b*S+s
      if (st == Sn - 1) hT[idx] = hv;
    }
#if HAVE_TDM
    if (w == 0) __builtin_amdgcn_s_wait_tensorcnt(0);      // next slab landed
#endif
    __syncthreads();
  }
}

// ---------------------------------------------------------------------------
// Decode: C[8192 x 32000] = H[8192x64] * Wd^T + bd, f16 WMMA, NT stores.
// Store-bandwidth bound (1.05 GB of logits @ 23.3 TB/s ~= 45us): NT stores
// keep the 192MB L2 from being flushed by a >1GB streaming output.
// ---------------------------------------------------------------------------
__global__ void __launch_bounds__(256)
decode(const _Float16* __restrict__ Hh, const _Float16* __restrict__ Wdh,
       const float* __restrict__ bd, float* __restrict__ out) {
  const int t = threadIdx.x, lane = t & 31, w = t >> 5;
  const int r0 = blockIdx.y * 128 + w * 16;
  const int v0 = blockIdx.x * 64;
  const int rl = lane & 15;
  const int hi = (lane & 16) ? 1 : 0;

  // A fragments (16x32 f16, contiguous K per half-wave) from row-major Hh
  const _Float16* ha = Hh + (size_t)(r0 + rl) * 64;
  int ab = hi ? 16 : 0;
  hvec8 a0lo = *(const hvec8*)(ha + ab);
  hvec8 a0hi = *(const hvec8*)(ha + ab + 8);
  hvec8 a1lo = *(const hvec8*)(ha + 32 + ab);
  hvec8 a1hi = *(const hvec8*)(ha + 32 + ab + 8);
  hvec16 a0 = __builtin_shufflevector(a0lo, a0hi, 0,1,2,3,4,5,6,7,8,9,10,11,12,13,14,15);
  hvec16 a1 = __builtin_shufflevector(a1lo, a1hi, 0,1,2,3,4,5,6,7,8,9,10,11,12,13,14,15);

#pragma unroll
  for (int vt = 0; vt < 4; ++vt) {
    int v = v0 + vt * 16 + rl;
    const _Float16* wb = Wdh + (size_t)v * 64 + (hi ? 16 : 0);
    hvec16 bf0 = *(const hvec16*)wb;          // K = n 0..31
    hvec16 bf1 = *(const hvec16*)(wb + 32);   // K = n 32..63
    fvec8 c = {0.f, 0.f, 0.f, 0.f, 0.f, 0.f, 0.f, 0.f};
    c = __builtin_amdgcn_wmma_f32_16x16x32_f16(false, a0, false, bf0, (short)0, c, false, false);
    c = __builtin_amdgcn_wmma_f32_16x16x32_f16(false, a1, false, bf1, (short)0, c, false, false);
    float bias = bd[v];
    int rowbase = r0 + (hi ? 8 : 0);
#pragma unroll
    for (int r = 0; r < 8; ++r) {
      float val = c[r] + bias;
      __builtin_nontemporal_store(val, out + (size_t)(rowbase + r) * Vn + v);
    }
  }
}

// ---------------------------------------------------------------------------
extern "C" void kernel_launch(void* const* d_in, const int* in_sizes, int n_in,
                              void* d_out, int out_size, void* d_ws, size_t ws_size,
                              hipStream_t stream) {
  const int*   ids = (const int*)d_in[0];
  const float* emb = (const float*)d_in[1];
  const float* W   = (const float*)d_in[2];
  const float* Wd  = (const float*)d_in[3];
  const float* bd  = (const float*)d_in[4];
  float* out = (float*)d_out;

  char* ws = (char*)d_ws;
  _Float16* Wp   = (_Float16*)ws;                                  // 524,288 B
  _Float16* Wdh  = (_Float16*)(ws + 524288);                       // 4,096,000 B
  _Float16* Hh   = (_Float16*)(ws + 524288 + 4096000);             // 1,048,576 B
  float*    word = (float*)(ws + 524288 + 4096000 + 1048576);      // 2,097,152 B

  pack_W<<<64, 256, 0, stream>>>(W, Wp);
  cvt_f16<<<(Vn * Nn + 255) / 256, 256, 0, stream>>>(Wd, Wdh, Vn * Nn);
  embed_relu<<<(Sn * Bn * Nn) / 256, 256, 0, stream>>>(ids, emb, word);
  recurrence<<<1, 512, LDS_TOTAL, stream>>>(Wp, word, Hh, out + (size_t)Bn * Sn * Vn);
  decode<<<dim3(Vn / 64, (Bn * Sn) / 128), 256, 0, stream>>>(Hh, Wdh, bd, out);
}
